// TransformerBlock_19387482374745
// MI455X (gfx1250) — matrix operand order repaired and checked by reading, not compiled, not run on hardware
//
#include <hip/hip_runtime.h>
#include <hip/hip_bf16.h>
#include <math.h>

// ---------------- problem constants ----------------
#define B_ 2
#define S_ 2048
#define D_ 1024
#define H_ 16
#define HD_ 64
#define FF_ 4096
#define M_ (B_ * S_)          // 4096 rows
#define SCALE_ 0.125f         // 1/sqrt(64)
#define EPS_ 1e-5f

typedef __bf16 bf16_t;
typedef __attribute__((ext_vector_type(16))) __bf16 v16bf;
typedef __attribute__((ext_vector_type(8)))  float  v8f;

union V16 { v16bf v; unsigned u[8]; };
union V8F { v8f  v; float    f[8]; };

// round-to-nearest-even f32 -> bf16 (bit pattern)
__device__ __forceinline__ unsigned short f2bf(float x) {
    unsigned u = __float_as_uint(x);
    return (unsigned short)((u + 0x7fffu + ((u >> 16) & 1u)) >> 16);
}
__device__ __forceinline__ unsigned pack2(float a, float b) {
    return (unsigned)f2bf(a) | ((unsigned)f2bf(b) << 16);
}
// ISA 7.12.2: 16-bit A-matrix 16x32 K pattern for vgpr v, lane-half
__device__ __forceinline__ int kpatA(int v, int half) {
    return (v < 4 ? 2 * v : 16 + 2 * (v - 4)) + 8 * half;
}
// 16-bit B-matrix 32x16 K pattern (same as A: lanes0-15: K0-7,16-23; lanes16-31: K8-15,24-31)
__device__ __forceinline__ int kpatB(int v, int half) {
    return (v < 4 ? 2 * v : 16 + 2 * (v - 4)) + 8 * half;
}

// ---------------- fp32 -> bf16 convert ----------------
__global__ void cvt_f32_bf16(const float* __restrict__ in,
                             unsigned short* __restrict__ out, int n) {
    int i = blockIdx.x * blockDim.x + threadIdx.x;
    if (i < n) out[i] = f2bf(in[i]);
}

// ---------------- generic bf16 WMMA GEMM ----------------
// C[M,N] = A[M,K] @ Bw[K,N]  (+bias) (+resid) (relu?) -> bf16 and/or f32
__launch_bounds__(256)
__global__ void gemm_bf16(const unsigned short* __restrict__ A,
                          const unsigned short* __restrict__ Bw,
                          const float* __restrict__ bias,
                          const float* __restrict__ resid,
                          unsigned short* __restrict__ outB,
                          float* __restrict__ outF,
                          int Mdim, int Ndim, int Kdim, int do_relu) {
    __shared__ unsigned short As[128][32];   // [m][k]
    __shared__ unsigned short Bs[128][32];   // transposed: [n][k]

    const int tid  = threadIdx.x;
    const int lane = tid & 31, wave = tid >> 5;
    const int half = lane >> 4, l = lane & 15;
    const int m0 = blockIdx.y * 128, n0 = blockIdx.x * 128;
    const int wm = (wave >> 1) * 32;   // 4 waves down
    const int wn = (wave & 1) * 64;    // 2 waves across

    V8F acc[2][4];
#pragma unroll
    for (int mi = 0; mi < 2; ++mi)
#pragma unroll
        for (int ni = 0; ni < 4; ++ni) acc[mi][ni].v = (v8f){};

    const unsigned* gA = (const unsigned*)A;
    const unsigned* gB = (const unsigned*)Bw;
    const int Ku = Kdim >> 1, Nu = Ndim >> 1;

    for (int k0 = 0; k0 < Kdim; k0 += 32) {
        __syncthreads();
        // stage A tile 128x32 (2048 dwords)
        unsigned* sA = (unsigned*)&As[0][0];
#pragma unroll
        for (int i = 0; i < 8; ++i) {
            int idx = i * 256 + tid;           // 0..2047
            int row = idx >> 4, cu = idx & 15; // 16 dwords per row
            sA[idx] = gA[(size_t)(m0 + row) * Ku + (k0 >> 1) + cu];
        }
        // stage B tile 32x128, transposed into Bs[n][k]
#pragma unroll
        for (int i = 0; i < 8; ++i) {
            int idx = i * 256 + tid;           // 0..2047
            int kr = idx >> 6, nu = idx & 63;  // 64 dwords per k-row
            unsigned w2 = gB[(size_t)(k0 + kr) * Nu + (n0 >> 1) + nu];
            Bs[2 * nu][kr]     = (unsigned short)(w2 & 0xffffu);
            Bs[2 * nu + 1][kr] = (unsigned short)(w2 >> 16);
        }
        __syncthreads();

        V16 af[2], bfr[4];
#pragma unroll
        for (int mi = 0; mi < 2; ++mi)
#pragma unroll
            for (int v = 0; v < 8; ++v)
                af[mi].u[v] = *(const unsigned*)&As[wm + mi * 16 + l][kpatA(v, half)];
#pragma unroll
        for (int ni = 0; ni < 4; ++ni)
#pragma unroll
            for (int v = 0; v < 8; ++v)
                bfr[ni].u[v] = *(const unsigned*)&Bs[wn + ni * 16 + l][kpatB(v, half)];

#pragma unroll
        for (int mi = 0; mi < 2; ++mi)
#pragma unroll
            for (int ni = 0; ni < 4; ++ni)
                acc[mi][ni].v = __builtin_amdgcn_wmma_f32_16x16x32_bf16(
                    false, af[mi].v, false, bfr[ni].v,
                    (short)0, acc[mi][ni].v, false, false);
    }

    // epilogue: C layout -> row = r + 8*half, col = lane&15
#pragma unroll
    for (int mi = 0; mi < 2; ++mi)
#pragma unroll
        for (int ni = 0; ni < 4; ++ni) {
            int col = n0 + wn + ni * 16 + l;
            float bsv = bias ? bias[col] : 0.f;
#pragma unroll
            for (int r = 0; r < 8; ++r) {
                int row = m0 + wm + mi * 16 + r + 8 * half;
                float vv = acc[mi][ni].f[r] + bsv;
                if (resid) vv += resid[(size_t)row * Ndim + col];
                if (do_relu) vv = fmaxf(vv, 0.f);
                if (outF) outF[(size_t)row * Ndim + col] = vv;
                if (outB) outB[(size_t)row * Ndim + col] = f2bf(vv);
            }
        }
}

// ---------------- flash attention (S^T = K*Q^T trick) ----------------
// qkv: bf16 [M][3*D], per (row,h): q at h*192+0, k at +64, v at +128
__launch_bounds__(128)
__global__ void attn_kernel(const unsigned short* __restrict__ qkv,
                            unsigned short* __restrict__ outB) {
    __shared__ unsigned short Ks[32][64];  // [key][hd]
    __shared__ unsigned short Vt[64][32];  // transposed [hd][key]

    const int tid  = threadIdx.x;
    const int lane = tid & 31, wave = tid >> 5;
    const int half = lane >> 4, l = lane & 15;
    const int bh = blockIdx.y;             // 0..31
    const int b = bh >> 4, h = bh & 15;
    const int qbase = blockIdx.x * 64 + wave * 16;
    const int rowbase = b * S_ + qbase;

    const unsigned* q32 = (const unsigned*)qkv;
    const int rs32 = (3 * D_) >> 1;        // dwords per row = 1536
    const int qo32 = (h * 192) >> 1;       // = h*96

    // preload Q^T B-fragments (lane = query column), 2 K-chunks of HD
    V16 bq[2];
#pragma unroll
    for (int c = 0; c < 2; ++c)
#pragma unroll
        for (int v = 0; v < 8; ++v) {
            int kk = kpatB(v, half) + c * 32;
            bq[c].u[v] = q32[(size_t)(rowbase + l) * rs32 + qo32 + (kk >> 1)];
        }

    V8F Oacc[4];
#pragma unroll
    for (int n = 0; n < 4; ++n) Oacc[n].v = (v8f){};
    float mrun = -1e30f, lrun = 0.f;

    for (int kt = 0; kt < S_; kt += 32) {
        __syncthreads();
        // stage K tile (32x64) and transposed V tile
#pragma unroll
        for (int i = 0; i < 8; ++i) {
            int idx = i * 128 + tid;            // 0..1023 dwords
            int r = idx >> 5, cu = idx & 31;    // 32 dwords per row
            ((unsigned*)&Ks[0][0])[r * 32 + cu] =
                q32[(size_t)(b * S_ + kt + r) * rs32 + qo32 + 32 + cu];
        }
#pragma unroll
        for (int i = 0; i < 8; ++i) {
            int idx = i * 128 + tid;
            int r = idx >> 5, cu = idx & 31;
            unsigned w2 = q32[(size_t)(b * S_ + kt + r) * rs32 + qo32 + 64 + cu];
            int d = 2 * cu;
            Vt[d][r]     = (unsigned short)(w2 & 0xffffu);
            Vt[d + 1][r] = (unsigned short)(w2 >> 16);
        }
        __syncthreads();

        // S^T tiles: (keys 0-15) and (keys 16-31) x (16 queries)
        V8F sc[2];
#pragma unroll
        for (int t = 0; t < 2; ++t) {
            v8f c = (v8f){};
#pragma unroll
            for (int cc = 0; cc < 2; ++cc) {
                V16 ak;
#pragma unroll
                for (int v = 0; v < 8; ++v)
                    ak.u[v] = *(const unsigned*)&Ks[t * 16 + l][kpatA(v, half) + cc * 32];
                c = __builtin_amdgcn_wmma_f32_16x16x32_bf16(
                    false, ak.v, false, bq[cc].v, (short)0, c, false, false);
            }
            sc[t].v = c;
        }

        // online softmax (per lane = per query; halves merged via xor16)
        float tmax = -1e30f;
#pragma unroll
        for (int t = 0; t < 2; ++t)
#pragma unroll
            for (int r = 0; r < 8; ++r) {
                sc[t].f[r] *= SCALE_;
                tmax = fmaxf(tmax, sc[t].f[r]);
            }
        tmax = fmaxf(tmax, __shfl_xor(tmax, 16, 32));
        float mnew = fmaxf(mrun, tmax);
        float fsc = __expf(mrun - mnew);
        mrun = mnew;
        float psum = 0.f;
#pragma unroll
        for (int t = 0; t < 2; ++t)
#pragma unroll
            for (int r = 0; r < 8; ++r) {
                float p = __expf(sc[t].f[r] - mnew);
                sc[t].f[r] = p;
                psum += p;
            }
        lrun = lrun * fsc + psum;

        // rescale output accumulators (broadcast factor per O-row)
        float fb[8];
#pragma unroll
        for (int r = 0; r < 8; ++r) fb[r] = __shfl(fsc, r + 8 * half, 32);
#pragma unroll
        for (int n = 0; n < 4; ++n)
#pragma unroll
            for (int r = 0; r < 8; ++r) Oacc[n].f[r] *= fb[r];

        // P^T C-layout maps straight onto the A-fragment layout
        V16 pa;
#pragma unroll
        for (int v = 0; v < 4; ++v) {
            pa.u[v]     = pack2(sc[0].f[2 * v], sc[0].f[2 * v + 1]);
            pa.u[4 + v] = pack2(sc[1].f[2 * v], sc[1].f[2 * v + 1]);
        }
        // O += P @ V  (4 HD tiles)
#pragma unroll
        for (int n = 0; n < 4; ++n) {
            V16 bv;
#pragma unroll
            for (int v = 0; v < 8; ++v)
                bv.u[v] = *(const unsigned*)&Vt[n * 16 + l][kpatB(v, half)];
            Oacc[n].v = __builtin_amdgcn_wmma_f32_16x16x32_bf16(
                false, pa.v, false, bv.v, (short)0, Oacc[n].v, false, false);
        }
    }

    // finalize: divide by row-sums, write bf16 [M][D], col = h*64 + d
    float ltot = lrun + __shfl_xor(lrun, 16, 32);
    float inv[8];
#pragma unroll
    for (int r = 0; r < 8; ++r) {
        float lt = __shfl(ltot, r + 8 * half, 32);
        inv[r] = 1.f / lt;
    }
#pragma unroll
    for (int n = 0; n < 4; ++n)
#pragma unroll
        for (int r = 0; r < 8; ++r) {
            int row = rowbase + r + 8 * half;
            int col = h * 64 + n * 16 + l;
            outB[(size_t)row * D_ + col] = f2bf(Oacc[n].f[r] * inv[r]);
        }
}

// ---------------- LayerNorm (one block per row, D=1024) ----------------
__launch_bounds__(256)
__global__ void ln_kernel(float* __restrict__ io,
                          const float* __restrict__ g,
                          const float* __restrict__ bt,
                          unsigned short* __restrict__ outB,   // nullable
                          float* __restrict__ outF) {          // null -> in place
    __shared__ float red[8];
    int row = blockIdx.x;
    const float* x = io + (size_t)row * D_;
    float v[4], s = 0.f;
#pragma unroll
    for (int i = 0; i < 4; ++i) { v[i] = x[threadIdx.x + i * 256]; s += v[i]; }
#pragma unroll
    for (int off = 16; off > 0; off >>= 1) s += __shfl_xor(s, off, 32);
    if ((threadIdx.x & 31) == 0) red[threadIdx.x >> 5] = s;
    __syncthreads();
    float tot = 0.f;
#pragma unroll
    for (int w = 0; w < 8; ++w) tot += red[w];
    float mean = tot * (1.0f / D_);
    float vs = 0.f;
#pragma unroll
    for (int i = 0; i < 4; ++i) { float d = v[i] - mean; vs += d * d; }
#pragma unroll
    for (int off = 16; off > 0; off >>= 1) vs += __shfl_xor(vs, off, 32);
    __syncthreads();
    if ((threadIdx.x & 31) == 0) red[threadIdx.x >> 5] = vs;
    __syncthreads();
    float vtot = 0.f;
#pragma unroll
    for (int w = 0; w < 8; ++w) vtot += red[w];
    float rstd = rsqrtf(vtot * (1.0f / D_) + EPS_);
    float* op = (outF ? outF : io) + (size_t)row * D_;
#pragma unroll
    for (int i = 0; i < 4; ++i) {
        int c = threadIdx.x + i * 256;
        float o = (v[i] - mean) * rstd * g[c] + bt[c];
        op[c] = o;
        if (outB) outB[(size_t)row * D_ + c] = f2bf(o);
    }
}

// ---------------- host orchestration ----------------
extern "C" void kernel_launch(void* const* d_in, const int* in_sizes, int n_in,
                              void* d_out, int out_size, void* d_ws, size_t ws_size,
                              hipStream_t stream) {
    const float* x     = (const float*)d_in[0];
    const float* w_qkv = (const float*)d_in[1];
    const float* w_o   = (const float*)d_in[2];
    const float* b_o   = (const float*)d_in[3];
    const float* g1    = (const float*)d_in[4];
    const float* bt1   = (const float*)d_in[5];
    const float* w1    = (const float*)d_in[6];
    const float* b1    = (const float*)d_in[7];
    const float* w2    = (const float*)d_in[8];
    const float* b2    = (const float*)d_in[9];
    const float* g2    = (const float*)d_in[10];
    const float* bt2   = (const float*)d_in[11];
    float* out = (float*)d_out;

    // workspace layout (256B aligned)
    char* ws = (char*)d_ws;
    size_t off = 0;
    auto alloc = [&](size_t bytes) {
        char* p = ws + off;
        off += (bytes + 255) & ~(size_t)255;
        return p;
    };
    unsigned short* xb    = (unsigned short*)alloc((size_t)M_ * D_ * 2);
    unsigned short* wqkvb = (unsigned short*)alloc((size_t)D_ * 3 * D_ * 2);
    unsigned short* wob   = (unsigned short*)alloc((size_t)D_ * D_ * 2);
    unsigned short* w1b   = (unsigned short*)alloc((size_t)D_ * FF_ * 2);
    unsigned short* w2b   = (unsigned short*)alloc((size_t)FF_ * D_ * 2);
    unsigned short* qkvb  = (unsigned short*)alloc((size_t)M_ * 3 * D_ * 2);
    unsigned short* aob   = (unsigned short*)alloc((size_t)M_ * D_ * 2);
    float*          ybuf  = (float*)alloc((size_t)M_ * D_ * 4);
    unsigned short* yb    = (unsigned short*)alloc((size_t)M_ * D_ * 2);
    unsigned short* hb    = (unsigned short*)alloc((size_t)M_ * FF_ * 2);

    auto cvt = [&](const float* src, unsigned short* dst, int n) {
        cvt_f32_bf16<<<(n + 255) / 256, 256, 0, stream>>>(src, dst, n);
    };
    cvt(x,     xb,    M_ * D_);
    cvt(w_qkv, wqkvb, D_ * 3 * D_);
    cvt(w_o,   wob,   D_ * D_);
    cvt(w1,    w1b,   D_ * FF_);
    cvt(w2,    w2b,   FF_ * D_);

    // 1) qkv = x @ w_qkv              [4096 x 3072], K=1024
    gemm_bf16<<<dim3(3 * D_ / 128, M_ / 128), 256, 0, stream>>>(
        xb, wqkvb, nullptr, nullptr, qkvb, nullptr, M_, 3 * D_, D_, 0);

    // 2) flash attention -> aob bf16 [4096 x 1024]
    attn_kernel<<<dim3(S_ / 64, B_ * H_), 128, 0, stream>>>(qkvb, aob);

    // 3) y_pre = x + ao @ w_o + b_o  -> ybuf f32
    gemm_bf16<<<dim3(D_ / 128, M_ / 128), 256, 0, stream>>>(
        aob, wob, b_o, x, nullptr, ybuf, M_, D_, D_, 0);

    // 4) y = LN1(y_pre) in place, + bf16 copy
    ln_kernel<<<M_, 256, 0, stream>>>(ybuf, g1, bt1, yb, nullptr);

    // 5) h = relu(y @ w1 + b1) -> hb bf16 [4096 x 4096], K=1024
    gemm_bf16<<<dim3(FF_ / 128, M_ / 128), 256, 0, stream>>>(
        yb, w1b, b1, nullptr, hb, nullptr, M_, FF_, D_, 1);

    // 6) out_pre = y + h @ w2 + b2 -> d_out f32, K=4096
    gemm_bf16<<<dim3(D_ / 128, M_ / 128), 256, 0, stream>>>(
        hb, w2b, b2, ybuf, nullptr, out, M_, D_, FF_, 0);

    // 7) out = LN2(out_pre) in place
    ln_kernel<<<M_, 256, 0, stream>>>(out, g2, bt2, nullptr, nullptr);
}
